// TrashCanSparseMoeBlock_26465588478517
// MI455X (gfx1250) — matrix operand order repaired and checked by reading, not compiled, not run on hardware
//
#include <hip/hip_runtime.h>

// ---- problem constants -----------------------------------------------------
#define HID   2048
#define FFN   1024
#define NTOK  8192          // 4 * 2048 tokens
#define NEXP  64
#define NTOT  72            // 64 + 8 trash experts
#define TOPK  8
#define TILE_M   32         // tokens per expert tile (2 WMMA M-tiles)
#define MAXTILES (NTOK / TILE_M)   // 256 (worst case: all tokens -> one expert)

typedef __bf16 v16bf __attribute__((ext_vector_type(16)));
typedef __bf16 v8bf  __attribute__((ext_vector_type(8)));
typedef __bf16 v4bf  __attribute__((ext_vector_type(4)));
typedef float  v8f   __attribute__((ext_vector_type(8)));

// ---- CDNA5 async global->LDS path (ASYNCcnt), with sync fallback -----------
#if defined(__has_builtin)
#if __has_builtin(__builtin_amdgcn_global_load_async_to_lds_b32) && \
    __has_builtin(__builtin_amdgcn_s_wait_asynccnt)
#define HAVE_ASYNC 1
#endif
#endif

#if defined(HAVE_ASYNC)
typedef __attribute__((address_space(1))) int gas_int;   // global (HIP prints as __device__ int*)
typedef __attribute__((address_space(3))) int las_int;   // LDS    (HIP prints as __shared__ int*)
#endif

__device__ __forceinline__ void copy_b32_to_lds(const float* __restrict__ g, float* l) {
#if defined(HAVE_ASYNC)
    __builtin_amdgcn_global_load_async_to_lds_b32((gas_int*)(void*)g, (las_int*)(void*)l, 0, 0);
#else
    *l = *g;
#endif
}

__device__ __forceinline__ void wait_async_lds() {
#if defined(HAVE_ASYNC)
    __builtin_amdgcn_s_wait_asynccnt(0);
#endif
}

// ---- helpers ---------------------------------------------------------------
// Build the 16-bit WMMA A operand for one lane from a bf16 row in LDS.
// Layout (contiguous k order, matching B): lanes 0-15 hold K = kb+[0..15];
// lanes 16-31 hold K = kb+[16..31].
__device__ __forceinline__ v16bf load_a_row(const __bf16* rowbase, int kb, int khalf) {
    const v8bf lo = *(const v8bf*)(rowbase + kb + khalf * 16);
    const v8bf hi = *(const v8bf*)(rowbase + kb + khalf * 16 + 8);
    return __builtin_shufflevector(lo, hi, 0,1,2,3,4,5,6,7,8,9,10,11,12,13,14,15);
}

// Convert 16 contiguous f32 (one lane's B column slice in LDS) to the bf16
// WMMA B operand: 4x ds_load_b128 + packed cvt.
__device__ __forceinline__ v16bf cvt_b(const float* __restrict__ p) {
    v16bf b;
    #pragma unroll
    for (int i = 0; i < 16; ++i) b[i] = (__bf16)p[i];
    return b;
}

// Issue one 32(k) x 128(col) f32 weight tile copy into LDS, pre-swizzled to
// the transposed [col][32] layout, via async b32 transfers (no VGPR roundtrip).
// LD is compile-time so global offsets fold into instruction immediates.
template<int LD>
__device__ __forceinline__ void issue_b_tile(const float* __restrict__ src, // &w[kb*LD + nb]
                                             float* __restrict__ dst,       // [128][32] f32
                                             int tid) {
    const int c  = tid & 127;           // column (coalesced across threads)
    const int r0 = tid >> 7;            // k phase: 0 or 1
    const float* g = src + (size_t)r0 * LD + c;
    float* l = dst + c * 32 + r0;
    #pragma unroll
    for (int i = 0; i < 16; ++i)        // k = r0 + 2i covers 0..31
        copy_b32_to_lds(g + (size_t)(2 * i) * LD, l + 2 * i);
}

// ---- zero kernels ----------------------------------------------------------
__global__ __launch_bounds__(256) void zero_out_kernel(float4* __restrict__ p) {
    p[(size_t)blockIdx.x * 256 + threadIdx.x] = make_float4(0.f, 0.f, 0.f, 0.f);
}

__global__ void zero_cnt_kernel(int* __restrict__ cnt) {
    if (threadIdx.x < NEXP) cnt[threadIdx.x] = 0;
}

// ---- router: logits[t, e] = x[t,:] . gate_w[e,:] ---------------------------
__global__ __launch_bounds__(256) void router_kernel(const float* __restrict__ x,
                                                     const float* __restrict__ gw,
                                                     float* __restrict__ logits) {
    const int wid  = threadIdx.x >> 5;
    const int lane = threadIdx.x & 31;
    const int t    = blockIdx.x * 8 + wid;             // 1024 blocks * 8 waves
    const float* xr = x + (size_t)t * HID;
    for (int e = 0; e < NTOT; ++e) {
        const float* gr = gw + (size_t)e * HID;
        float s = 0.f;
        for (int h = lane; h < HID; h += 32) s += xr[h] * gr[h];
        #pragma unroll
        for (int off = 16; off > 0; off >>= 1) s += __shfl_down(s, off, 32);
        if (lane == 0) logits[(size_t)t * NTOT + e] = s;
    }
}

// ---- top-k softmax routing; scatter (token, weight) into per-expert lists --
__global__ __launch_bounds__(256) void topk_kernel(const float* __restrict__ logits,
                                                   int*   __restrict__ cnt,
                                                   int*   __restrict__ etok,
                                                   float* __restrict__ ew) {
    const int t = blockIdx.x * 256 + threadIdx.x;      // 32 blocks -> 8192 tokens
    const float* lr = logits + (size_t)t * NTOT;
    float l[NTOT];
    float mx = -1e30f;
    for (int e = 0; e < NTOT; ++e) { l[e] = lr[e]; mx = fmaxf(mx, l[e]); }
    float den = 0.f;
    for (int e = 0; e < NTOT; ++e) den += __expf(l[e] - mx);
    const float inv = __builtin_amdgcn_rcpf(den);
    for (int k = 0; k < TOPK; ++k) {
        int bi = 0; float bv = -1e30f;
        for (int e = 0; e < NTOT; ++e) if (l[e] > bv) { bv = l[e]; bi = e; }
        l[bi] = -1e30f;
        if (bi < NEXP) {                                // trash experts dropped
            const float w = __expf(bv - mx) * inv;      // no renorm (norm_topk_prob=False)
            const int pos = atomicAdd(&cnt[bi], 1);
            etok[bi * NTOK + pos] = t;
            ew  [bi * NTOK + pos] = w;
        }
    }
}

// ---- fused expert FFN: one workgroup = (expert, 32-token tile) -------------
// 8 waves / 256 threads. bf16 WMMA 16x16x32, f32 accumulate. B tiles are
// double-buffered in LDS and filled by async global->LDS DMA (ASYNCcnt),
// pipelined against the WMMA burst; one barrier per 32-K step.
__global__ __launch_bounds__(256) void moe_ffn_kernel(const float* __restrict__ x,
                                                      const float* __restrict__ wg,
                                                      const float* __restrict__ wu,
                                                      const float* __restrict__ wd,
                                                      const int*   __restrict__ cnt,
                                                      const int*   __restrict__ etok,
                                                      const float* __restrict__ ew,
                                                      float* __restrict__ out) {
    const int e    = blockIdx.x >> 8;                  // / MAXTILES
    const int tile = blockIdx.x & (MAXTILES - 1);
    const int count = cnt[e];
    if (tile * TILE_M >= count) return;

    extern __shared__ char smem[];
    __bf16* Xs = (__bf16*)smem;                              // [32][2048] 128 KB
    __bf16* Is = (__bf16*)(smem + TILE_M * HID * 2);         // [32][1024]  64 KB
    float*  BF = (float*)(smem + (size_t)TILE_M * (HID + FFN) * 2); // 4x[128][32] f32, 64 KB
    __shared__ int   tok[TILE_M];
    __shared__ float wt[TILE_M];

    const int tid   = threadIdx.x;
    const int wid   = tid >> 5;
    const int lane  = tid & 31;
    const int col   = lane & 15;        // N column (B/C/D) == M row (A)
    const int khalf = lane >> 4;

    if (tid < TILE_M) {
        const int pos = tile * TILE_M + tid;
        if (pos < count) { tok[tid] = etok[e * NTOK + pos]; wt[tid] = ew[e * NTOK + pos]; }
        else             { tok[tid] = etok[e * NTOK + tile * TILE_M]; wt[tid] = 0.f; }
    }
    __syncthreads();

    // gather 32 token rows of x into LDS as bf16 (float4 loads, v4bf stores)
    {
        const float4* x4 = (const float4*)x;
        #pragma unroll 4
        for (int i = 0; i < 64; ++i) {
            const int idx = tid + (i << 8);            // 0..16383 float4 slots
            const int r   = idx >> 9;                  // / 512
            const int c4  = idx & 511;
            const float4 v = x4[(size_t)tok[r] * (HID / 4) + c4];
            v4bf p; p[0] = (__bf16)v.x; p[1] = (__bf16)v.y;
                    p[2] = (__bf16)v.z; p[3] = (__bf16)v.w;
            *(v4bf*)(Xs + (r << 11) + (c4 << 2)) = p;
        }
    }

    const float* wgp = wg + (size_t)e * HID * FFN;     // gate: [H][F]
    const float* wup = wu + (size_t)e * HID * FFN;     // up:   [H][F]
    const float* wdp = wd + (size_t)e * FFN * HID;     // down: [F][H]
    const int blane = (wid * 16 + col) * 32 + khalf * 16;

    // ---- phase 1: G = X*Wg, U = X*Wu, I = silu(G)*U ------------------------
    for (int nb = 0; nb < FFN; nb += 128) {            // 8 waves x 16 cols
        v8f aG0 = {}, aG1 = {}, aU0 = {}, aU1 = {};
        __syncthreads();                               // B region free to refill
        issue_b_tile<FFN>(wgp + nb, BF,        tid);   // gate buf0
        issue_b_tile<FFN>(wup + nb, BF + 8192, tid);   // up   buf0
        for (int kb = 0; kb < HID; kb += 32) {
            const int buf = (kb >> 5) & 1;
            wait_async_lds();
            __syncthreads();                           // cur tile ready, prev reads done
            if (kb + 32 < HID) {                       // prefetch next k-block via DMA
                issue_b_tile<FFN>(wgp + (size_t)(kb + 32) * FFN + nb, BF + (buf ^ 1) * 4096, tid);
                issue_b_tile<FFN>(wup + (size_t)(kb + 32) * FFN + nb, BF + 8192 + (buf ^ 1) * 4096, tid);
            }
            const v16bf a0 = load_a_row(Xs + col * HID, kb, khalf);
            const v16bf a1 = load_a_row(Xs + (col + 16) * HID, kb, khalf);
            const v16bf bg = cvt_b(BF + buf * 4096 + blane);
            const v16bf bu = cvt_b(BF + 8192 + buf * 4096 + blane);
            aG0 = __builtin_amdgcn_wmma_f32_16x16x32_bf16(false, a0, false, bg, (short)0, aG0, false, false);
            aG1 = __builtin_amdgcn_wmma_f32_16x16x32_bf16(false, a1, false, bg, (short)0, aG1, false, false);
            aU0 = __builtin_amdgcn_wmma_f32_16x16x32_bf16(false, a0, false, bu, (short)0, aU0, false, false);
            aU1 = __builtin_amdgcn_wmma_f32_16x16x32_bf16(false, a1, false, bu, (short)0, aU1, false, false);
        }
        // D layout: VGPR j -> row (j + 8*khalf), col = lane&15
        #pragma unroll
        for (int j = 0; j < 8; ++j) {
            const int r = j + khalf * 8;
            const int fcol = nb + wid * 16 + col;
            float g0 = aG0[j];
            float s0 = g0 * __builtin_amdgcn_rcpf(1.f + __expf(-g0));   // silu
            Is[r * FFN + fcol] = (__bf16)(s0 * aU0[j]);
            float g1 = aG1[j];
            float s1 = g1 * __builtin_amdgcn_rcpf(1.f + __expf(-g1));
            Is[(r + 16) * FFN + fcol] = (__bf16)(s1 * aU1[j]);
        }
    }

    // ---- phase 2: Out += (I * Wd) * routing_weight -------------------------
    for (int nb = 0; nb < HID; nb += 128) {
        v8f acc0 = {}, acc1 = {};
        __syncthreads();                               // covers Is writes + B reuse
        issue_b_tile<HID>(wdp + nb, BF, tid);
        for (int kb = 0; kb < FFN; kb += 32) {
            const int buf = (kb >> 5) & 1;
            wait_async_lds();
            __syncthreads();
            if (kb + 32 < FFN)
                issue_b_tile<HID>(wdp + (size_t)(kb + 32) * HID + nb, BF + (buf ^ 1) * 4096, tid);
            const v16bf a0 = load_a_row(Is + col * FFN, kb, khalf);
            const v16bf a1 = load_a_row(Is + (col + 16) * FFN, kb, khalf);
            const v16bf b  = cvt_b(BF + buf * 4096 + blane);
            acc0 = __builtin_amdgcn_wmma_f32_16x16x32_bf16(false, a0, false, b, (short)0, acc0, false, false);
            acc1 = __builtin_amdgcn_wmma_f32_16x16x32_bf16(false, a1, false, b, (short)0, acc1, false, false);
        }
        #pragma unroll
        for (int j = 0; j < 8; ++j) {
            const int r = j + khalf * 8;
            const int hcol = nb + wid * 16 + col;
            atomicAdd(&out[(size_t)tok[r] * HID + hcol],      wt[r] * acc0[j]);
            atomicAdd(&out[(size_t)tok[r + 16] * HID + hcol], wt[r + 16] * acc1[j]);
        }
    }
}

// ---- launch ----------------------------------------------------------------
extern "C" void kernel_launch(void* const* d_in, const int* in_sizes, int n_in,
                              void* d_out, int out_size, void* d_ws, size_t ws_size,
                              hipStream_t stream) {
    const float* x   = (const float*)d_in[0];   // [4,2048,2048]
    const float* gw  = (const float*)d_in[1];   // [72,2048]
    const float* wgt = (const float*)d_in[2];   // [64,2048,1024]
    const float* wup = (const float*)d_in[3];   // [64,2048,1024]
    const float* wdn = (const float*)d_in[4];   // [64,1024,2048]

    float* out    = (float*)d_out;                       // 16,777,216 f32
    float* logits = out + (size_t)NTOK * HID;            // 8192*72 f32

    char* ws    = (char*)d_ws;
    int*   cnt  = (int*)ws;                              // 64 ints
    int*   etok = (int*)(ws + 256);                      // 64*8192 ints (2 MB)
    float* ew   = (float*)(ws + 256 + (size_t)NEXP * NTOK * 4); // 2 MB

    // 1. zero accumulator + counters
    zero_out_kernel<<<(NTOK * HID) / (4 * 256), 256, 0, stream>>>((float4*)out);
    zero_cnt_kernel<<<1, 64, 0, stream>>>(cnt);

    // 2. router logits (also an output)
    router_kernel<<<NTOK / 8, 256, 0, stream>>>(x, gw, logits);

    // 3. softmax + top-8 routing, build per-expert token lists
    topk_kernel<<<NTOK / 256, 256, 0, stream>>>(logits, cnt, etok, ew);

    // 4. fused expert FFN (bf16 WMMA + async LDS staging), expert-ordered grid
    const size_t lds_bytes = (size_t)TILE_M * (HID + FFN) * 2 + 4 * 128 * 32 * 4; // 262,144 B
    moe_ffn_kernel<<<NEXP * MAXTILES, 256, lds_bytes, stream>>>(
        x, wgt, wup, wdn, cnt, etok, ew, out);
}
